// GPT_72018011619708
// MI455X (gfx1250) — matrix operand order repaired and checked by reading, not compiled, not run on hardware
//
#include <hip/hip_runtime.h>
#include <cstdint>
#include <cmath>

typedef __bf16 bf16_t;
typedef __attribute__((ext_vector_type(4)))  bf16_t bf16x4;
typedef __attribute__((ext_vector_type(8)))  bf16_t bf16x8;
typedef __attribute__((ext_vector_type(16))) bf16_t bf16x16;
typedef __attribute__((ext_vector_type(8)))  float  f32x8;
typedef __attribute__((ext_vector_type(4)))  unsigned int u32x4;
typedef __attribute__((ext_vector_type(8)))  int i32x8;
typedef __attribute__((ext_vector_type(4)))  int i32x4;

constexpr int BATCH = 2;
constexpr int T     = 1024;
constexpr int Cdim  = 1024;
constexpr int NH    = 16;
constexpr int HD    = 64;     // head dim
constexpr int NL    = 4;
constexpr int VOCAB = 50257;

#if __has_builtin(__builtin_amdgcn_tensor_load_to_lds) && __has_builtin(__builtin_amdgcn_s_wait_tensorcnt)
#define USE_TDM 1
#else
#define USE_TDM 0
#endif

__device__ __forceinline__ bf16x16 cat8(bf16x8 lo, bf16x8 hi) {
  return __builtin_shufflevector(lo, hi, 0,1,2,3,4,5,6,7,8,9,10,11,12,13,14,15);
}

__device__ __forceinline__ f32x8 wmma_bf16(bf16x16 a, bf16x16 b, f32x8 c) {
  // D = A(16x32) * B(32x16) + C, f32 accumulate
  return __builtin_amdgcn_wmma_f32_16x16x32_bf16(false, a, false, b, (short)0, c, false, false);
}

#if USE_TDM
// ---------------------------------------------------------------------------
// TDM 2D tile load: Global -> LDS, one DMA op, tracked by TENSORcnt.
// D# packed per CDNA5 ISA 8.3/8.4:
//  g0: [1:0]=count=1, [63:32]=lds_addr, [120:64]=global_addr, [127:126]=type=2
//  g1: [17:16]=data_size, [79:48]=tensor_dim0, [111:80]=tensor_dim1,
//      [127:112]=tile_dim0, [143:128]=tile_dim1, [207:160]=tensor_dim0_stride
// OOB rows (y >= tensor_dim1) read as zero.
// ---------------------------------------------------------------------------
__device__ __forceinline__ void tdm_load_2d(
    unsigned lds_off, const void* gptr, unsigned data_size_code /*1=2B,2=4B*/,
    unsigned tile_w, unsigned tile_h, unsigned tensor_w, unsigned tensor_h,
    unsigned long long row_stride_elems)
{
  unsigned long long ga = (unsigned long long)(uintptr_t)gptr;
  u32x4 g0;
  g0[0] = 1u;                                               // count=1 (user D#)
  g0[1] = lds_off;                                          // lds_addr (bytes)
  g0[2] = (unsigned)(ga & 0xFFFFFFFFu);                     // global_addr[31:0]
  g0[3] = (unsigned)((ga >> 32) & 0x01FFFFFFu) | (2u << 30); // addr[56:32] | type=2
  i32x8 g1;
  g1[0] = (int)(data_size_code << 16);                      // wg_mask=0, data_size
  g1[1] = (int)((tensor_w & 0xFFFFu) << 16);                // tensor_dim0[15:0]
  g1[2] = (int)((tensor_w >> 16) | ((tensor_h & 0xFFFFu) << 16));
  g1[3] = (int)((tensor_h >> 16) | (tile_w << 16));         // tile_dim0
  g1[4] = (int)(tile_h);                                    // tile_dim1, tile_dim2=0
  g1[5] = (int)(row_stride_elems & 0xFFFFFFFFull);          // dim0_stride[31:0]
  g1[6] = (int)((row_stride_elems >> 32) & 0xFFFFull);      // dim0_stride[47:32]
  g1[7] = 0;
  i32x4 z4; z4[0] = 0; z4[1] = 0; z4[2] = 0; z4[3] = 0;
#if __clang_major__ >= 23
  i32x8 z8; z8[0]=0; z8[1]=0; z8[2]=0; z8[3]=0; z8[4]=0; z8[5]=0; z8[6]=0; z8[7]=0;
  __builtin_amdgcn_tensor_load_to_lds(g0, g1, z4, z4, z8, 0);
#else
  __builtin_amdgcn_tensor_load_to_lds(g0, g1, z4, z4, 0);
#endif
}
#endif

// ---------------------------------------------------------------------------
// Embedding: X[row, c] = tok_emb[idx[row], c] + pos_emb[row % T, c]   (fp32)
// ---------------------------------------------------------------------------
__global__ __launch_bounds__(256) void embed_kernel(
    const int* __restrict__ idx, const float* __restrict__ tok,
    const float* __restrict__ pos, float* __restrict__ X)
{
  const int row = blockIdx.x;
  const int t   = row & (T - 1);
  const int token = idx[row];
  const int c = threadIdx.x * 4;
  float4 a = *reinterpret_cast<const float4*>(&tok[(size_t)token * Cdim + c]);
  float4 p = *reinterpret_cast<const float4*>(&pos[(size_t)t * Cdim + c]);
  float4 o; o.x = a.x + p.x; o.y = a.y + p.y; o.z = a.z + p.z; o.w = a.w + p.w;
  *reinterpret_cast<float4*>(&X[(size_t)row * Cdim + c]) = o;
}

// ---------------------------------------------------------------------------
// LayerNorm over C=1024, one row per 256-thread block, bf16 output
// ---------------------------------------------------------------------------
__global__ __launch_bounds__(256) void layernorm_kernel(
    const float* __restrict__ X, const float* __restrict__ w,
    const float* __restrict__ bvec, bf16_t* __restrict__ out)
{
  __shared__ float red[16];
  const int row = blockIdx.x;
  const int tid = threadIdx.x;
  const int c = tid * 4;
  float4 v = *reinterpret_cast<const float4*>(&X[(size_t)row * Cdim + c]);
  float s = v.x + v.y + v.z + v.w;
  float q = v.x*v.x + v.y*v.y + v.z*v.z + v.w*v.w;
  #pragma unroll
  for (int mm = 16; mm >= 1; mm >>= 1) { s += __shfl_xor(s, mm, 32); q += __shfl_xor(q, mm, 32); }
  if ((tid & 31) == 0) { red[tid >> 5] = s; red[8 + (tid >> 5)] = q; }
  __syncthreads();
  s = 0.f; q = 0.f;
  #pragma unroll
  for (int wv = 0; wv < 8; ++wv) { s += red[wv]; q += red[8 + wv]; }
  const float mu  = s * (1.0f / 1024.0f);
  const float var = q * (1.0f / 1024.0f) - mu * mu;
  const float rs  = rsqrtf(var + 1e-5f);
  bf16x4 o;
  o[0] = (bf16_t)((v.x - mu) * rs * w[c + 0] + bvec[c + 0]);
  o[1] = (bf16_t)((v.y - mu) * rs * w[c + 1] + bvec[c + 1]);
  o[2] = (bf16_t)((v.z - mu) * rs * w[c + 2] + bvec[c + 2]);
  o[3] = (bf16_t)((v.w - mu) * rs * w[c + 3] + bvec[c + 3]);
  *reinterpret_cast<bf16x4*>(&out[(size_t)row * Cdim + c]) = o;
}

// ---------------------------------------------------------------------------
// NT GEMM: out[m,n] = epilogue( sum_k A[m,k]*W[n,k] )
//   A: bf16 [M,K] row-major, W: fp32 [N,K] row-major.
//   Tiles staged into LDS by the Tensor Data Mover (double-buffered); wave 0
//   issues both TDM descriptors, s_wait_tensorcnt + barrier publishes them.
//   MODE 0: out bf16 = acc + bias
//   MODE 1: out f32  = acc + bias + res[m,n]
//   MODE 2: out bf16 = gelu(acc + bias)          (exact, erf-based)
//   MODE 3: out f32  = acc                       (head; OOB rows zero-filled
//                                                 by TDM, store guarded)
// Block: 256 thr (8 waves, 4x2), tile 128x128x32, each wave 32x64 = 8 WMMAs.
// ---------------------------------------------------------------------------
template <int MODE>
__global__ __launch_bounds__(256) void gemm_nt(
    const bf16_t* __restrict__ A, const float* __restrict__ W,
    const float* __restrict__ bias, const float* __restrict__ res,
    void* __restrict__ outp, int M, int N, int K)
{
  __shared__ bf16_t As[2][128][32];    // 16 KB
  __shared__ float  Bs[2][128][32];    // 32 KB (fp32 weights, cvt at frag build)
  const int tid  = threadIdx.x;
  const int lane = tid & 31, wid = tid >> 5;
  const int ln   = lane & 15, hb = lane >> 4;
  const int wm   = wid & 3,   wn = wid >> 2;
  const int m0   = blockIdx.y * 128, n0 = blockIdx.x * 128;

  f32x8 acc[2][4];
  #pragma unroll
  for (int i = 0; i < 2; ++i)
    #pragma unroll
    for (int j = 0; j < 4; ++j)
      #pragma unroll
      for (int e = 0; e < 8; ++e) acc[i][j][e] = 0.0f;

  const int nk = K >> 5;

#if USE_TDM
  const unsigned ldsA0 = (unsigned)(uintptr_t)&As[0][0][0];
  const unsigned ldsA1 = (unsigned)(uintptr_t)&As[1][0][0];
  const unsigned ldsB0 = (unsigned)(uintptr_t)&Bs[0][0][0];
  const unsigned ldsB1 = (unsigned)(uintptr_t)&Bs[1][0][0];
  if (wid == 0) {   // prologue: stage k-slice 0 into buffer 0
    tdm_load_2d(ldsA0, &A[(size_t)m0 * K], 1u, 32u, 128u, 32u, 128u,
                (unsigned long long)K);
    tdm_load_2d(ldsB0, &W[(size_t)n0 * K], 2u, 32u, 128u, 32u, (unsigned)(N - n0),
                (unsigned long long)K);
  }
#endif

  for (int kb = 0; kb < nk; ++kb) {
    const int buf = kb & 1;
#if USE_TDM
    if (wid == 0) __builtin_amdgcn_s_wait_tensorcnt(0);
    __syncthreads();   // tiles[buf] ready; prior reads of tiles[buf^1] done
    if (wid == 0 && kb + 1 < nk) {
      const unsigned la = (buf == 0) ? ldsA1 : ldsA0;
      const unsigned lb = (buf == 0) ? ldsB1 : ldsB0;
      tdm_load_2d(la, &A[(size_t)m0 * K + (kb + 1) * 32], 1u, 32u, 128u, 32u, 128u,
                  (unsigned long long)K);
      tdm_load_2d(lb, &W[(size_t)n0 * K + (kb + 1) * 32], 2u, 32u, 128u, 32u,
                  (unsigned)(N - n0), (unsigned long long)K);
    }
#else
    __syncthreads();
    // Fallback: manual staging (A: 512 16B chunks / B: 1024 float4 chunks)
    #pragma unroll
    for (int i = 0; i < 2; ++i) {
      int c = tid + i * 256;
      int row = c >> 2, col = (c & 3) << 3;
      *reinterpret_cast<bf16x8*>(&As[buf][row][col]) =
          *reinterpret_cast<const bf16x8*>(&A[(size_t)(m0 + row) * K + kb * 32 + col]);
    }
    #pragma unroll
    for (int i = 0; i < 4; ++i) {
      int c = tid + i * 256;
      int row = c >> 3, col = (c & 7) << 2;
      int gn = n0 + row; if (gn > N - 1) gn = N - 1;
      *reinterpret_cast<float4*>(&Bs[buf][row][col]) =
          *reinterpret_cast<const float4*>(&W[(size_t)gn * K + kb * 32 + col]);
    }
    __syncthreads();
#endif

    // Fragments: contiguous k order (k = 16*half + s), same for A and B
    bf16x16 af[2], bfr[4];
    #pragma unroll
    for (int i = 0; i < 2; ++i) {        // A 16x32: lane row = ln, K = 16*half..+15
      int row = wm * 32 + i * 16 + ln;
      int k0  = hb * 16;
      bf16x8 lo = *reinterpret_cast<const bf16x8*>(&As[buf][row][k0]);
      bf16x8 hi = *reinterpret_cast<const bf16x8*>(&As[buf][row][k0 + 8]);
      af[i] = cat8(lo, hi);
    }
    #pragma unroll
    for (int j = 0; j < 4; ++j) {        // B 32x16: lane col = ln, K = 16*half..+15, cvt f32->bf16
      int row = wn * 64 + j * 16 + ln;
      int k0  = hb * 16;
      const float4* bp = reinterpret_cast<const float4*>(&Bs[buf][row][k0]);
      float4 f0 = bp[0], f1 = bp[1], f2 = bp[2], f3 = bp[3];
      bf16x16 bb;
      bb[0]  = (bf16_t)f0.x; bb[1]  = (bf16_t)f0.y; bb[2]  = (bf16_t)f0.z; bb[3]  = (bf16_t)f0.w;
      bb[4]  = (bf16_t)f1.x; bb[5]  = (bf16_t)f1.y; bb[6]  = (bf16_t)f1.z; bb[7]  = (bf16_t)f1.w;
      bb[8]  = (bf16_t)f2.x; bb[9]  = (bf16_t)f2.y; bb[10] = (bf16_t)f2.z; bb[11] = (bf16_t)f2.w;
      bb[12] = (bf16_t)f3.x; bb[13] = (bf16_t)f3.y; bb[14] = (bf16_t)f3.z; bb[15] = (bf16_t)f3.w;
      bfr[j] = bb;
    }
    #pragma unroll
    for (int i = 0; i < 2; ++i)
      #pragma unroll
      for (int j = 0; j < 4; ++j)
        acc[i][j] = wmma_bf16(af[i], bfr[j], acc[i][j]);
  }

  // Epilogue. C/D layout: vgpr r -> m = r + 8*half, n = ln
  #pragma unroll
  for (int i = 0; i < 2; ++i) {
    #pragma unroll
    for (int j = 0; j < 4; ++j) {
      const int n = n0 + wn * 64 + j * 16 + ln;
      if (n >= N) continue;
      const float bb = (MODE == 3) ? 0.0f : bias[n];
      #pragma unroll
      for (int r = 0; r < 8; ++r) {
        const int m = m0 + wm * 32 + i * 16 + hb * 8 + r;
        float v = acc[i][j][r] + bb;
        if (MODE == 1) v += res[(size_t)m * N + n];
        if (MODE == 2) v = 0.5f * v * (1.0f + erff(v * 0.70710678118654752440f));
        if (MODE == 0 || MODE == 2)
          reinterpret_cast<bf16_t*>(outp)[(size_t)m * N + n] = (bf16_t)v;
        else
          reinterpret_cast<float*>(outp)[(size_t)m * N + n] = v;
      }
    }
  }
}

// ---------------------------------------------------------------------------
// Causal flash attention, one wave per (b, h, 16-row q tile), D=64.
// S = Q K^T via 4 WMMAs per 32-key block, online softmax in C-fragment
// registers, P(16x32)*V(32x64) via 4 WMMAs (P restaged through LDS to get
// the A-fragment layout).
// ---------------------------------------------------------------------------
__global__ __launch_bounds__(32) void attn_kernel(
    const bf16_t* __restrict__ Q, const bf16_t* __restrict__ Kp,
    const bf16_t* __restrict__ Vp, bf16_t* __restrict__ Y)
{
  __shared__ bf16_t Vs[32][64];   // staged V keys
  __shared__ bf16_t Ps[16][32];   // P re-layout buffer
  const int lane = threadIdx.x;
  const int ln = lane & 15, hb = lane >> 4;
  const int q0 = blockIdx.x * 16;
  const int h  = blockIdx.y;
  const int b  = blockIdx.z;
  const float scale = 0.125f;     // 1/sqrt(64)

  // Q fragments: rows q0..q0+15, K-dim = head channels (2 k-steps of 32)
  bf16x16 aq[2];
  {
    const bf16_t* qrow = Q + ((size_t)(b * T + q0 + ln)) * Cdim + h * HD + hb * 8;
    #pragma unroll
    for (int kk = 0; kk < 2; ++kk) {
      bf16x8 lo = *reinterpret_cast<const bf16x8*>(qrow + kk * 32);
      bf16x8 hi = *reinterpret_cast<const bf16x8*>(qrow + kk * 32 + 16);
      aq[kk] = cat8(lo, hi);
    }
  }

  f32x8 O[4];
  #pragma unroll
  for (int t = 0; t < 4; ++t)
    #pragma unroll
    for (int e = 0; e < 8; ++e) O[t][e] = 0.0f;
  float mreg[8], lreg[8];
  #pragma unroll
  for (int r = 0; r < 8; ++r) { mreg[r] = -INFINITY; lreg[r] = 0.0f; }

  const int nb = ((q0 + 15) >> 5) + 1;   // 32-key blocks needed
  for (int jb = 0; jb < nb; ++jb) {
    const int kbase = jb * 32;

    // Stage V block: lane loads one key row (64 bf16 = 8x16B)
    {
      const bf16_t* vrow = Vp + ((size_t)(b * T + kbase + lane)) * Cdim + h * HD;
      #pragma unroll
      for (int c = 0; c < 8; ++c)
        *reinterpret_cast<bf16x8*>(&Vs[lane][c * 8]) =
            *reinterpret_cast<const bf16x8*>(vrow + c * 8);
    }

    // S = Q K^T for this 16x32 block (two 16x16 n-subtiles, 2 k-steps each)
    f32x8 S[2];
    #pragma unroll
    for (int nt = 0; nt < 2; ++nt) {
      #pragma unroll
      for (int e = 0; e < 8; ++e) S[nt][e] = 0.0f;
      #pragma unroll
      for (int kk = 0; kk < 2; ++kk) {
        const bf16_t* krow = Kp + ((size_t)(b * T + kbase + nt * 16 + ln)) * Cdim
                                + h * HD + kk * 32 + hb * 8;
        bf16x8 lo = *reinterpret_cast<const bf16x8*>(krow);
        bf16x8 hi = *reinterpret_cast<const bf16x8*>(krow + 16);
        S[nt] = wmma_bf16(aq[kk], cat8(lo, hi), S[nt]);
      }
    }
    __syncthreads();

    // Online softmax: row m = r + 8*half lives on the 16 lanes of this half
    #pragma unroll
    for (int r = 0; r < 8; ++r) {
      const int mrow = hb * 8 + r;
      const int qg = q0 + mrow;
      float v0 = S[0][r] * scale; if (kbase + ln      > qg) v0 = -INFINITY;
      float v1 = S[1][r] * scale; if (kbase + 16 + ln > qg) v1 = -INFINITY;
      float bm = fmaxf(v0, v1);
      #pragma unroll
      for (int mm = 8; mm >= 1; mm >>= 1) bm = fmaxf(bm, __shfl_xor(bm, mm, 32));
      const float nm = fmaxf(mreg[r], bm);
      const float p0 = __expf(v0 - nm), p1 = __expf(v1 - nm);
      float ps = p0 + p1;
      #pragma unroll
      for (int mm = 8; mm >= 1; mm >>= 1) ps += __shfl_xor(ps, mm, 32);
      const float f = __expf(mreg[r] - nm);
      lreg[r] = lreg[r] * f + ps;
      mreg[r] = nm;
      #pragma unroll
      for (int t = 0; t < 4; ++t) O[t][r] *= f;
      Ps[mrow][ln]      = (bf16_t)p0;
      Ps[mrow][16 + ln] = (bf16_t)p1;
    }
    __syncthreads();

    // O += P(16x32) * V(32x64)
    bf16x16 ap;
    {
      const int k0 = hb * 8;
      bf16x8 lo = *reinterpret_cast<const bf16x8*>(&Ps[ln][k0]);
      bf16x8 hi = *reinterpret_cast<const bf16x8*>(&Ps[ln][k0 + 16]);
      ap = cat8(lo, hi);
    }
    #pragma unroll
    for (int t = 0; t < 4; ++t) {
      bf16x16 bvf;
      const int d = t * 16 + ln;
      #pragma unroll
      for (int e = 0; e < 16; ++e) bvf[e] = Vs[hb * 8 + e + (e >= 8 ? 8 : 0)][d];   // B col d, keys 8*half..+7, 16+8*half..+7
      O[t] = wmma_bf16(ap, bvf, O[t]);
    }
    __syncthreads();
  }

  // Normalize and write y[b, q, h*64 + d] (bf16)
  #pragma unroll
  for (int t = 0; t < 4; ++t) {
    #pragma unroll
    for (int r = 0; r < 8; ++r) {
      const int mrow = hb * 8 + r;
      const float o = O[t][r] / lreg[r];
      Y[((size_t)(b * T + q0 + mrow)) * Cdim + h * HD + t * 16 + ln] = (bf16_t)o;
    }
  }
}

// ---------------------------------------------------------------------------
extern "C" void kernel_launch(void* const* d_in, const int* in_sizes, int n_in,
                              void* d_out, int out_size, void* d_ws, size_t ws_size,
                              hipStream_t stream) {
  (void)in_sizes; (void)n_in; (void)out_size; (void)ws_size;
  const int*   idx   = (const int*)  d_in[0];
  const float* tok   = (const float*)d_in[1];
  const float* pos   = (const float*)d_in[2];
  const float* ln1w  = (const float*)d_in[3];
  const float* ln1b  = (const float*)d_in[4];
  const float* Wq    = (const float*)d_in[5];
  const float* bq    = (const float*)d_in[6];
  const float* Wk    = (const float*)d_in[7];
  const float* bk    = (const float*)d_in[8];
  const float* Wv    = (const float*)d_in[9];
  const float* bvp   = (const float*)d_in[10];
  const float* Wo    = (const float*)d_in[11];
  const float* bo    = (const float*)d_in[12];
  const float* ln2w  = (const float*)d_in[13];
  const float* ln2b  = (const float*)d_in[14];
  const float* W1    = (const float*)d_in[15];
  const float* b1    = (const float*)d_in[16];
  const float* W2    = (const float*)d_in[17];
  const float* b2    = (const float*)d_in[18];
  const float* lnfw  = (const float*)d_in[19];
  const float* lnfb  = (const float*)d_in[20];
  const float* headw = (const float*)d_in[21];

  const int M = BATCH * T;  // 2048
  // Workspace layout (~44 MB)
  float*  X  = (float*)d_ws;                           // M*C fp32 residual
  bf16_t* Hc = (bf16_t*)(X + (size_t)M * Cdim);        // M*C bf16 (LN out)
  bf16_t* Qb = Hc + (size_t)M * Cdim;
  bf16_t* Kb = Qb + (size_t)M * Cdim;
  bf16_t* Vb = Kb + (size_t)M * Cdim;
  bf16_t* Yb = Vb + (size_t)M * Cdim;
  bf16_t* H4 = Yb + (size_t)M * Cdim;                  // M*4C bf16 (MLP hidden)

  const dim3 blk256(256), blk32(32);
  const dim3 gC(Cdim / 128, M / 128);          // 8 x 16
  const dim3 gF(4 * Cdim / 128, M / 128);      // 32 x 16
  const dim3 gHead((VOCAB + 127) / 128, M / 128);
  const dim3 gAttn(T / 16, NH, BATCH);

  embed_kernel<<<dim3(M), blk256, 0, stream>>>(idx, tok, pos, X);

  for (int l = 0; l < NL; ++l) {
    const size_t oC = (size_t)l * Cdim, oCC = (size_t)l * Cdim * Cdim;
    const size_t oF = (size_t)l * 4 * Cdim, oFC = (size_t)l * 4 * Cdim * Cdim;
    layernorm_kernel<<<dim3(M), blk256, 0, stream>>>(X, ln1w + oC, ln1b + oC, Hc);
    gemm_nt<0><<<gC, blk256, 0, stream>>>(Hc, Wq + oCC, bq + oC, nullptr, Qb, M, Cdim, Cdim);
    gemm_nt<0><<<gC, blk256, 0, stream>>>(Hc, Wk + oCC, bk + oC, nullptr, Kb, M, Cdim, Cdim);
    gemm_nt<0><<<gC, blk256, 0, stream>>>(Hc, Wv + oCC, bvp + oC, nullptr, Vb, M, Cdim, Cdim);
    attn_kernel<<<gAttn, blk32, 0, stream>>>(Qb, Kb, Vb, Yb);
    gemm_nt<1><<<gC, blk256, 0, stream>>>(Yb, Wo + oCC, bo + oC, X, X, M, Cdim, Cdim);
    layernorm_kernel<<<dim3(M), blk256, 0, stream>>>(X, ln2w + oC, ln2b + oC, Hc);
    gemm_nt<2><<<gF, blk256, 0, stream>>>(Hc, W1 + oFC, b1 + oF, nullptr, H4, M, 4 * Cdim, Cdim);
    gemm_nt<1><<<gC, blk256, 0, stream>>>(H4, W2 + oFC, b2 + oC, X, X, M, Cdim, 4 * Cdim);
  }

  layernorm_kernel<<<dim3(M), blk256, 0, stream>>>(X, lnfw, lnfb, Hc);
  gemm_nt<3><<<gHead, blk256, 0, stream>>>(Hc, headw, nullptr, nullptr, d_out, M, VOCAB, Cdim);
}
